// MaskRCNN_33621003993833
// MI455X (gfx1250) — compile-verified, matrix operand order checked by reading, not run on hardware
//
#include <hip/hip_runtime.h>
#include <hip/hip_bf16.h>
#include <math.h>

// ---- problem constants (match reference) ----
#define RR 512
#define KK 81
#define IMW_M1 1332.0f   // IM_W - 1
#define IMH_M1 799.0f    // IM_H - 1
#define XFORM_CLIP 4.135166556742356f  // log(1000/16)
#define NMS_T 0.5f
#define SCORE_T 0.001f
#define TOPN 300
#define MAXPER 100

typedef _Float16 v16h __attribute__((ext_vector_type(16)));
typedef float    v8f  __attribute__((ext_vector_type(8)));

// ---------------------------------------------------------------------------
// Kernel 1: bbox_transform -> boxes3[r][k][4]
// ---------------------------------------------------------------------------
__global__ void bbox_kernel(const float* __restrict__ rois,
                            const float* __restrict__ deltas,
                            float* __restrict__ boxes3) {
  int idx = blockIdx.x * blockDim.x + threadIdx.x;
  if (idx >= RR * KK) return;
  int r = idx / KK;
  int k = idx - r * KK;
  float x1 = rois[r * 4 + 0], y1 = rois[r * 4 + 1];
  float x2 = rois[r * 4 + 2], y2 = rois[r * 4 + 3];
  float w  = x2 - x1 + 1.0f;
  float h  = y2 - y1 + 1.0f;
  float cx = x1 + 0.5f * w;
  float cy = y1 + 0.5f * h;
  const float* d = deltas + (size_t)r * (4 * KK) + 4 * k;
  float dx = d[0] / 10.0f;
  float dy = d[1] / 10.0f;
  float dw = fminf(d[2] / 5.0f, XFORM_CLIP);
  float dh = fminf(d[3] / 5.0f, XFORM_CLIP);
  float pcx = dx * w + cx;
  float pcy = dy * h + cy;
  float pw  = expf(dw) * w;
  float ph  = expf(dh) * h;
  float ox1 = fminf(fmaxf(pcx - 0.5f * pw, 0.0f), IMW_M1);
  float oy1 = fminf(fmaxf(pcy - 0.5f * ph, 0.0f), IMH_M1);
  float ox2 = fminf(fmaxf(pcx + 0.5f * pw - 1.0f, 0.0f), IMW_M1);
  float oy2 = fminf(fmaxf(pcy + 0.5f * ph - 1.0f, 0.0f), IMH_M1);
  float* o = boxes3 + (size_t)idx * 4;
  o[0] = ox1; o[1] = oy1; o[2] = ox2; o[3] = oy2;
}

// ---------------------------------------------------------------------------
// Kernel 2: per-class NMS (one workgroup per class, all in LDS)
//   - bitonic sort 512 entries by (score desc, idx asc)  == stable argsort(-s)
//   - 512x512 IoU suppression bitmask (16 u32 words per row)
//   - sequential greedy scan (thread 0, registers)
//   - POST_NMS_TOPN cap: chunk totals via WMMA (exact: 0/1 f16 -> f32 acc),
//     chunk-offset prefix via wave32 __shfl, short LDS scans for positions
//   - scatter dists[k][orig] = keep * score
// ---------------------------------------------------------------------------
__global__ void __launch_bounds__(512)
nms_class_kernel(const float* __restrict__ scores,
                 const float* __restrict__ boxes3,
                 float* __restrict__ dists) {
  const int k   = blockIdx.x;
  const int tid = threadIdx.x;

  __shared__ float    s_sc[RR];
  __shared__ int      s_id[RR];
  __shared__ float    s_bx[RR][4];
  __shared__ float    s_ar[RR];
  __shared__ float    s_red[RR];
  __shared__ unsigned s_sup[RR][16];
  __shared__ unsigned s_keepw[16];
  __shared__ unsigned char s_keepb[RR];
  __shared__ float    s_choff[16];

  float sc = scores[tid * KK + k];
  s_sc[tid] = sc;
  s_id[tid] = tid;
  s_red[tid] = sc;
  __syncthreads();

  // class validity: k>0 and max score over rois > SCORE_T
  for (int s = 256; s > 0; s >>= 1) {
    if (tid < s) s_red[tid] = fmaxf(s_red[tid], s_red[tid + s]);
    __syncthreads();
  }
  bool cls_valid = (k != 0) && (s_red[0] > SCORE_T);
  if (!cls_valid) {                     // uniform branch across the block
    dists[k * RR + tid] = 0.0f;
    return;
  }

  // ---- bitonic sort: (score desc, idx asc) strict total order ----
  for (int len = 2; len <= RR; len <<= 1) {
    for (int j = len >> 1; j > 0; j >>= 1) {
      int p = tid ^ j;
      if (p > tid) {
        bool asc = ((tid & len) == 0);
        float sa = s_sc[tid], sb = s_sc[p];
        int   ia = s_id[tid], ib = s_id[p];
        bool b_first = (sb > sa) || ((sb == sa) && (ib < ia));
        bool dosw = asc ? b_first : !b_first;
        if (dosw) {
          s_sc[tid] = sb; s_sc[p] = sa;
          s_id[tid] = ib; s_id[p] = ia;
        }
      }
      __syncthreads();
    }
  }

  // ---- gather sorted boxes + areas ----
  {
    int orig = s_id[tid];
    const float* bp = boxes3 + ((size_t)orig * KK + k) * 4;
    float bx1 = bp[0], by1 = bp[1], bx2 = bp[2], by2 = bp[3];
    s_bx[tid][0] = bx1; s_bx[tid][1] = by1;
    s_bx[tid][2] = bx2; s_bx[tid][3] = by2;
    s_ar[tid] = (bx2 - bx1 + 1.0f) * (by2 - by1 + 1.0f);
  }
  __syncthreads();

  // ---- suppression bitmask row: bit j set iff j>i and IoU(i,j) > 0.5 ----
  {
    float ai  = s_ar[tid];
    float bx1 = s_bx[tid][0], by1 = s_bx[tid][1];
    float bx2 = s_bx[tid][2], by2 = s_bx[tid][3];
    for (int w = 0; w < 16; ++w) {
      unsigned bits = 0u;
      int jb = w << 5;
      for (int jj = 0; jj < 32; ++jj) {
        int j = jb + jj;
        if (j > tid) {
          float xx1 = fmaxf(bx1, s_bx[j][0]);
          float yy1 = fmaxf(by1, s_bx[j][1]);
          float xx2 = fminf(bx2, s_bx[j][2]);
          float yy2 = fminf(by2, s_bx[j][3]);
          float iw = fmaxf(xx2 - xx1 + 1.0f, 0.0f);
          float ih = fmaxf(yy2 - yy1 + 1.0f, 0.0f);
          float inter = iw * ih;
          float iou = inter / (ai + s_ar[j] - inter);
          if (iou > NMS_T) bits |= (1u << jj);
        }
      }
      s_sup[tid][w] = bits;
    }
  }
  __syncthreads();

  // ---- greedy scan (inherently sequential) ----
  if (tid == 0) {
    unsigned kw[16];
    for (int w = 0; w < 16; ++w) kw[w] = 0xFFFFFFFFu;
    for (int i = 0; i < RR; ++i) {
      if ((kw[i >> 5] >> (i & 31)) & 1u) {
        for (int w = 0; w < 16; ++w) kw[w] &= ~s_sup[i][w];
      }
    }
    for (int w = 0; w < 16; ++w) s_keepw[w] = kw[w];
  }
  __syncthreads();
  s_keepb[tid] = (unsigned char)((s_keepw[tid >> 5] >> (tid & 31)) & 1u);
  __syncthreads();

  // ---- TOPN cap. Chunk totals (16 chunks of 32) via one WMMA:
  //      A = ones(16x32), B[k][n] = keep[n*32+k]  => D[m][n] = total(chunk n).
  //      Exact: 0/1 in f16, f32 accumulate; permutation-invariant in K. ----
  if (tid < 32) {                       // wave 0: EXEC all ones for WMMA
    int l  = tid;
    int hi = l >> 4;
    int n  = l & 15;
    v16h a, bm;
#pragma unroll
    for (int s = 0; s < 16; ++s) {
      a[s]  = (_Float16)1.0f;
      bm[s] = (_Float16)(s_keepb[n * 32 + s + 16 * hi] ? 1.0f : 0.0f);
    }
    v8f c = {};
    v8f d = __builtin_amdgcn_wmma_f32_16x16x32_f16(
        /*neg_a=*/false, a, /*neg_b=*/false, bm,
        /*c_mod=*/(short)0, c, /*reuse_a=*/false, /*reuse_b=*/false);
    // D layout: VGPR0, lanes 0..15 -> (M=0, N=lane); lanes 16..31 -> (M=8, N=lane-16)
    float tot = d[0];                   // total of chunk (l & 15)
    float off = 0.0f;                   // exclusive prefix across chunks
#pragma unroll
    for (int c2 = 0; c2 < 15; ++c2) {
      float t = __shfl(tot, c2, 32);
      if (c2 < n) off += t;
    }
    if (l < 16) s_choff[l] = off;
  }
  __syncthreads();

  // inclusive cumsum at position tid = chunk offset + local scan, cap at 300
  {
    int chunk = tid >> 5;
    float cum = s_choff[chunk];
    int base = chunk << 5;
    for (int p = base; p <= tid; ++p) cum += (float)s_keepb[p];
    unsigned char newk =
        (s_keepb[tid] && (cum <= (float)TOPN)) ? (unsigned char)1 : (unsigned char)0;
    __syncthreads();                    // all reads of s_keepb done before writes
    s_keepb[tid] = newk;
    // scatter to original roi index
    dists[k * RR + s_id[tid]] = newk ? s_sc[tid] : 0.0f;
  }
}

// ---------------------------------------------------------------------------
// Kernel 3: per-roi max / first-argmax over classes
// ---------------------------------------------------------------------------
__global__ void rowmax_kernel(const float* __restrict__ dists,
                              float* __restrict__ spre,
                              int* __restrict__ lpre) {
  int r = blockIdx.x * blockDim.x + threadIdx.x;
  if (r >= RR) return;
  float best = dists[r];                // k = 0
  int bk = 0;
  for (int k = 1; k < KK; ++k) {
    float v = dists[k * RR + r];
    if (v > best) { best = v; bk = k; } // strict > => first occurrence
  }
  spre[r] = best;
  lpre[r] = bk;
}

// ---------------------------------------------------------------------------
// Kernel 4: global top-100 (stable desc sort), assemble outputs
//   d_out layout (700 floats): out[100][5] | labels[100] | top[100]
// ---------------------------------------------------------------------------
__global__ void __launch_bounds__(512)
final_kernel(const float* __restrict__ spre,
             const int* __restrict__ lpre,
             const float* __restrict__ boxes3,
             float* __restrict__ out) {
  const int tid = threadIdx.x;
  __shared__ float s_sc[RR];
  __shared__ int   s_id[RR];
  s_sc[tid] = spre[tid];
  s_id[tid] = tid;
  __syncthreads();
  for (int len = 2; len <= RR; len <<= 1) {
    for (int j = len >> 1; j > 0; j >>= 1) {
      int p = tid ^ j;
      if (p > tid) {
        bool asc = ((tid & len) == 0);
        float sa = s_sc[tid], sb = s_sc[p];
        int   ia = s_id[tid], ib = s_id[p];
        bool b_first = (sb > sa) || ((sb == sa) && (ib < ia));
        bool dosw = asc ? b_first : !b_first;
        if (dosw) {
          s_sc[tid] = sb; s_sc[p] = sa;
          s_id[tid] = ib; s_id[p] = ia;
        }
      }
      __syncthreads();
    }
  }
  if (tid < MAXPER) {
    int top   = s_id[tid];
    float scv = s_sc[tid];
    int lab   = lpre[top];
    bool valid = scv > SCORE_T;
    const float* bp = boxes3 + ((size_t)top * KK + lab) * 4;
    out[tid * 5 + 0] = valid ? scv : 0.0f;
    out[tid * 5 + 1] = valid ? bp[0] : 0.0f;
    out[tid * 5 + 2] = valid ? bp[1] : 0.0f;
    out[tid * 5 + 3] = valid ? bp[2] : 0.0f;
    out[tid * 5 + 4] = valid ? bp[3] : 0.0f;
    out[500 + tid] = (float)lab;        // labels_all (unmasked, per reference)
    out[600 + tid] = (float)top;        // top (unmasked, per reference)
  }
}

// ---------------------------------------------------------------------------
extern "C" void kernel_launch(void* const* d_in, const int* in_sizes, int n_in,
                              void* d_out, int out_size, void* d_ws, size_t ws_size,
                              hipStream_t stream) {
  const float* rois   = (const float*)d_in[0];   // (512, 4)
  const float* deltas = (const float*)d_in[1];   // (512, 324)
  const float* scores = (const float*)d_in[2];   // (512, 81)
  float* out = (float*)d_out;                    // 700 floats

  // workspace layout (~834 KB total)
  float* boxes3 = (float*)d_ws;                  // 512*81*4 floats
  float* dists  = boxes3 + (size_t)RR * KK * 4;  // 81*512 floats (class-major)
  float* spre   = dists + (size_t)KK * RR;       // 512 floats
  int*   lpre   = (int*)(spre + RR);             // 512 ints

  bbox_kernel<<<(RR * KK + 255) / 256, 256, 0, stream>>>(rois, deltas, boxes3);
  nms_class_kernel<<<KK, RR, 0, stream>>>(scores, boxes3, dists);
  rowmax_kernel<<<2, 256, 0, stream>>>(dists, spre, lpre);
  final_kernel<<<1, RR, 0, stream>>>(spre, lpre, boxes3, out);
}
